// SCAM_39917426049713
// MI455X (gfx1250) — matrix operand order repaired and checked by reading, not compiled, not run on hardware
//
#include <hip/hip_runtime.h>

// ---------------------------------------------------------------------------
// SCAM stereo cross-attention for gfx1250 (MI455X).
// One workgroup per (b,h): 1024 blocks x 256 threads (8 wave32 waves).
// All GEMMs via v_wmma_f32_16x16x32_f16; LN/softmax/epilogue in f32.
// ---------------------------------------------------------------------------

typedef _Float16 h4  __attribute__((ext_vector_type(4)));
typedef _Float16 h8  __attribute__((ext_vector_type(8)));
typedef _Float16 h16 __attribute__((ext_vector_type(16)));
typedef float    f8  __attribute__((ext_vector_type(8)));

#define CCH   96      // channels
#define WW    256     // spatial width per (b,h)
#define NROW  (8 * 128 * 256)  // total rows per tensor

// LDS layout (in _Float16 units)
#define OFF_QL   0
#define OFF_QR   24576
#define OFF_VLT  49152            // V_l transposed: [c][w]
#define OFF_VRT  73728            // V_r transposed: [c][w]
#define OFF_STG  98304            // 49152 halfs: phase1 ln+raw / phase2 P tiles
#define OFF_WGT  147456           // 9216 halfs
#define SMEM_HALFS 156672         // 313344 bytes

__device__ __forceinline__ f8 wmma_f16(h16 a, h16 b, f8 c) {
  return __builtin_amdgcn_wmma_f32_16x16x32_f16(false, a, false, b, (short)0, c,
                                                false, false);
}

// A fragment, 16x32 f16, src row-major with leading dim ld (halfs).
// Lane m(0..15)=row; lanes>=16 carry K+8. Halfs j0..7 -> K=[k0,k0+8),
// j8..15 -> K=[k0+16,k0+24) with k0 = (lane>=16)*8.  Two aligned b128 loads.
__device__ __forceinline__ h16 load_a_frag(const _Float16* src, int ld,
                                           int row0, int kBase, int lane) {
  int m  = row0 + (lane & 15);
  int k0 = kBase + ((lane >> 4) << 3);
  const _Float16* p = src + m * ld + k0;
  h8 lo = *(const h8*)(p);
  h8 hi = *(const h8*)(p + 16);
  return __builtin_shufflevector(lo, hi, 0, 1, 2, 3, 4, 5, 6, 7,
                                 8, 9, 10, 11, 12, 13, 14, 15);
}

// B fragment, 32x16 f16, element (k,n) at src[n*ld + k] (n-major storage).
// Lane n(0..15)=col; halfs j0..7 -> K=[k0,k0+8), j8..15 -> K=[k0+16,k0+24), k0=(lane>=16)*8.
__device__ __forceinline__ h16 load_b_frag(const _Float16* src, int ld,
                                           int col0, int kBase, int lane) {
  int n  = col0 + (lane & 15);
  int k0 = kBase + ((lane >> 4) << 3);
  const _Float16* p = src + n * ld + k0;
  h8 lo = *(const h8*)(p);
  h8 hi = *(const h8*)(p + 16);
  return __builtin_shufflevector(lo, hi, 0, 1, 2, 3, 4, 5, 6, 7,
                                 8, 9, 10, 11, 12, 13, 14, 15);
}

// Stage one side: per-thread row LayerNorm -> f16 into lnDst, raw f16 -> rawDst.
__device__ __forceinline__ void stage_side(const float* __restrict__ x,
                                           const float* __restrict__ g,
                                           const float* __restrict__ b,
                                           _Float16* lnDst, _Float16* rawDst) {
  const int t = threadIdx.x;                 // one row (spatial pos) per thread
  const float4* row = (const float4*)(x + (size_t)t * CCH);
  float4 v[24];
  float s = 0.f, s2 = 0.f;
#pragma unroll
  for (int i = 0; i < 24; ++i) {
    float4 q = row[i];
    v[i] = q;
    s  += q.x + q.y + q.z + q.w;
    s2 += q.x * q.x + q.y * q.y + q.z * q.z + q.w * q.w;
  }
  const float inv96 = 1.0f / 96.0f;
  float mu  = s * inv96;
  float var = s2 * inv96 - mu * mu;
  float rs  = rsqrtf(var + 1e-6f);
  const float4* g4 = (const float4*)g;
  const float4* b4 = (const float4*)b;
  _Float16* lnp  = lnDst  + t * CCH;
  _Float16* rawp = rawDst + t * CCH;
#pragma unroll
  for (int i = 0; i < 24; ++i) {
    float4 q = v[i], gg = g4[i], bb = b4[i];
    lnp[i * 4 + 0] = (_Float16)((q.x - mu) * rs * gg.x + bb.x);
    lnp[i * 4 + 1] = (_Float16)((q.y - mu) * rs * gg.y + bb.y);
    lnp[i * 4 + 2] = (_Float16)((q.z - mu) * rs * gg.z + bb.z);
    lnp[i * 4 + 3] = (_Float16)((q.w - mu) * rs * gg.w + bb.w);
    rawp[i * 4 + 0] = (_Float16)q.x;
    rawp[i * 4 + 1] = (_Float16)q.y;
    rawp[i * 4 + 2] = (_Float16)q.z;
    rawp[i * 4 + 3] = (_Float16)q.w;
  }
}

__device__ __forceinline__ void load_wgt(const float* __restrict__ w,
                                         _Float16* wgt) {
  // W[d][c] row-major f32 -> f16 as-is; B access pattern reads wgt[n*96+k].
  const float4* w4 = (const float4*)w;
  for (int i = threadIdx.x; i < (CCH * CCH) / 4; i += 256) {
    float4 q = w4[i];
    h4 p = { (_Float16)q.x, (_Float16)q.y, (_Float16)q.z, (_Float16)q.w };
    *(h4*)&wgt[i * 4] = p;
  }
}

// Projection GEMM: out[256,96] = A[256,96] @ W^T, +bias, *scl.
// TRANS=false: store row-major [w][c]; TRANS=true: store transposed [c][w].
template <bool TRANS>
__device__ __forceinline__ void proj_gemm(const _Float16* aStage,
                                          const _Float16* wgt,
                                          const float* __restrict__ bias,
                                          float scl, _Float16* dst) {
  const int lane = threadIdx.x & 31;
  const int wv   = threadIdx.x >> 5;
  const int hi8  = (lane >> 4) << 3;
#pragma unroll 1
  for (int i = 0; i < 12; ++i) {
    int tile = wv * 12 + i;                 // 96 output tiles / 8 waves
    int mT = tile / 6, nT = tile % 6;
    f8 acc = {};
#pragma unroll
    for (int kk = 0; kk < 3; ++kk) {
      h16 a = load_a_frag(aStage, CCH, mT * 16, kk * 32, lane);
      h16 b = load_b_frag(wgt, CCH, nT * 16, kk * 32, lane);
      acc = wmma_f16(a, b, acc);
    }
    int col = nT * 16 + (lane & 15);
    float bv = bias[col];
    if (TRANS) {
      h8 pk;
#pragma unroll
      for (int r = 0; r < 8; ++r) pk[r] = (_Float16)((acc[r] + bv) * scl);
      *(h8*)&dst[col * WW + mT * 16 + hi8] = pk;   // [c][w], 16B aligned
    } else {
#pragma unroll
      for (int r = 0; r < 8; ++r)
        dst[(mT * 16 + hi8 + r) * CCH + col] = (_Float16)((acc[r] + bv) * scl);
    }
  }
}

// One attention direction: S = Qa @ Qb^T (scale prefolded into Q_l),
// row-softmax, F = P @ V; out = x + F * gvec.  Vt is [c][w].
__device__ __forceinline__ void attend(const _Float16* Qa, const _Float16* Qb,
                                       const _Float16* Vt,
                                       const float* __restrict__ x,
                                       const float* __restrict__ gvec,
                                       float* __restrict__ out,
                                       _Float16* pbuf) {
  const int lane = threadIdx.x & 31;
  const int wv   = threadIdx.x >> 5;
  const int hi8  = (lane >> 4) << 3;
  _Float16* P = pbuf + wv * (16 * WW);      // wave-private 8KB P tile
#pragma unroll 1
  for (int blk = wv; blk < 16; blk += 8) {
    const int mBase = blk * 16;
    // ---- scores: 16 x 256 in registers (16 C-fragments) ----
    h16 aq[3];
#pragma unroll
    for (int kk = 0; kk < 3; ++kk)
      aq[kk] = load_a_frag(Qa, CCH, mBase, kk * 32, lane);
    f8 acc[16];
#pragma unroll
    for (int f = 0; f < 16; ++f) {
      f8 c = {};
#pragma unroll
      for (int kk = 0; kk < 3; ++kk) {
        h16 b = load_b_frag(Qb, CCH, f * 16, kk * 32, lane);
        c = wmma_f16(aq[kk], b, c);
      }
      acc[f] = c;
    }
    // ---- row softmax: in-lane over 16 frags, xor-shuffle over 16 lanes ----
    float inv[8];
#pragma unroll
    for (int r = 0; r < 8; ++r) {
      float mx = acc[0][r];
#pragma unroll
      for (int f = 1; f < 16; ++f) mx = fmaxf(mx, acc[f][r]);
      mx = fmaxf(mx, __shfl_xor(mx, 1));
      mx = fmaxf(mx, __shfl_xor(mx, 2));
      mx = fmaxf(mx, __shfl_xor(mx, 4));
      mx = fmaxf(mx, __shfl_xor(mx, 8));
      float sum = 0.f;
#pragma unroll
      for (int f = 0; f < 16; ++f) {
        float e = __expf(acc[f][r] - mx);
        acc[f][r] = e;
        sum += e;
      }
      sum += __shfl_xor(sum, 1);
      sum += __shfl_xor(sum, 2);
      sum += __shfl_xor(sum, 4);
      sum += __shfl_xor(sum, 8);
      inv[r] = 1.0f / sum;
    }
    // ---- re-layout P (C-layout -> row-major LDS -> A fragments) ----
#pragma unroll
    for (int f = 0; f < 16; ++f)
#pragma unroll
      for (int r = 0; r < 8; ++r)
        P[(hi8 + r) * WW + f * 16 + (lane & 15)] = (_Float16)acc[f][r];
    h16 ap[8];
#pragma unroll
    for (int kk = 0; kk < 8; ++kk)
      ap[kk] = load_a_frag(P, WW, 0, kk * 32, lane);
    // ---- PV: [16x256] @ [256x96] ----
#pragma unroll 1
    for (int n = 0; n < 6; ++n) {
      f8 o = {};
#pragma unroll
      for (int kk = 0; kk < 8; ++kk) {
        h16 b = load_b_frag(Vt, WW, n * 16, kk * 32, lane);
        o = wmma_f16(ap[kk], b, o);
      }
      int c = n * 16 + (lane & 15);
      float g = gvec[c];
#pragma unroll
      for (int r = 0; r < 8; ++r) {
        int w = mBase + hi8 + r;
        out[(size_t)w * CCH + c] = x[(size_t)w * CCH + c] + o[r] * inv[r] * g;
      }
    }
  }
}

__global__ __launch_bounds__(256, 1) void scam_attn_kernel(
    const float* __restrict__ fl, const float* __restrict__ fr,
    const float* __restrict__ gnl, const float* __restrict__ bnl,
    const float* __restrict__ gnr, const float* __restrict__ bnr,
    const float* __restrict__ wl1, const float* __restrict__ bl1,
    const float* __restrict__ wr1, const float* __restrict__ br1,
    const float* __restrict__ wl2, const float* __restrict__ bl2,
    const float* __restrict__ wr2, const float* __restrict__ br2,
    const float* __restrict__ beta, const float* __restrict__ gamma,
    float* __restrict__ outL, float* __restrict__ outR) {
  __shared__ __align__(16) _Float16 smem[SMEM_HALFS];
  _Float16* qL  = smem + OFF_QL;
  _Float16* qR  = smem + OFF_QR;
  _Float16* vLt = smem + OFF_VLT;
  _Float16* vRt = smem + OFF_VRT;
  _Float16* stg = smem + OFF_STG;            // ln at +0, raw at +24576
  _Float16* wgt = smem + OFF_WGT;

  const int bh = blockIdx.x;                 // b*H + h, 0..1023
  const size_t base = (size_t)bh * WW * CCH;
  const float* xl = fl + base;
  const float* xr = fr + base;
  const float qscale = 0.10206207261596575f; // 96^-0.5, folded into Q_l

  // ---------------- phase 1: projections ----------------
  stage_side(xl, gnl, bnl, stg, stg + 24576);
  load_wgt(wl1, wgt);
  __syncthreads();
  proj_gemm<false>(stg, wgt, bl1, qscale, qL);          // Q_l (scaled)
  __syncthreads();
  load_wgt(wl2, wgt);
  __syncthreads();
  proj_gemm<true>(stg + 24576, wgt, bl2, 1.0f, vLt);    // V_l^T
  __syncthreads();
  stage_side(xr, gnr, bnr, stg, stg + 24576);
  load_wgt(wr1, wgt);
  __syncthreads();
  proj_gemm<false>(stg, wgt, br1, 1.0f, qR);            // Q_r
  __syncthreads();
  load_wgt(wr2, wgt);
  __syncthreads();
  proj_gemm<true>(stg + 24576, wgt, br2, 1.0f, vRt);    // V_r^T
  __syncthreads();

  // ---------------- phase 2: both attention directions ----------------
  attend(qL, qR, vRt, xl, beta,  outL + base, stg);     // r2l
  attend(qR, qL, vLt, xr, gamma, outR + base, stg);     // l2r (S^T symmetric)
}

extern "C" void kernel_launch(void* const* d_in, const int* in_sizes, int n_in,
                              void* d_out, int out_size, void* d_ws,
                              size_t ws_size, hipStream_t stream) {
  const float* fl  = (const float*)d_in[0];
  const float* fr  = (const float*)d_in[1];
  const float* gnl = (const float*)d_in[2];
  const float* bnl = (const float*)d_in[3];
  const float* gnr = (const float*)d_in[4];
  const float* bnr = (const float*)d_in[5];
  const float* wl1 = (const float*)d_in[6];
  const float* bl1 = (const float*)d_in[7];
  const float* wr1 = (const float*)d_in[8];
  const float* br1 = (const float*)d_in[9];
  const float* wl2 = (const float*)d_in[10];
  const float* bl2 = (const float*)d_in[11];
  const float* wr2 = (const float*)d_in[12];
  const float* br2 = (const float*)d_in[13];
  const float* beta  = (const float*)d_in[14];
  const float* gamma = (const float*)d_in[15];

  float* outL = (float*)d_out;
  float* outR = outL + (size_t)NROW * CCH;

  dim3 grid(1024), block(256);
  scam_attn_kernel<<<grid, block, 0, stream>>>(
      fl, fr, gnl, bnl, gnr, bnr, wl1, bl1, wr1, br1, wl2, bl2, wr2, br2,
      beta, gamma, outL, outR);
}
